// MultiHeadAttention_25245817766393
// MI455X (gfx1250) — matrix operand order repaired and checked by reading, not compiled, not run on hardware
//
#include <hip/hip_runtime.h>

typedef _Float16 f16;
typedef __attribute__((ext_vector_type(2)))  _Float16 v2h;
typedef __attribute__((ext_vector_type(8)))  _Float16 v8h;
typedef __attribute__((ext_vector_type(16))) _Float16 v16h;
typedef __attribute__((ext_vector_type(8)))  float    v8f;

#define B_   4
#define S_   4096
#define D_   64
#define LDH  72   // padded LDS row stride in halves (16B-aligned rows)

// softmax(s*scale + mask*NEG), scale = 1/sqrt(4096) = 1/64, computed as
// exp2 in the RAW score domain: p = exp2((s_raw + mask*NEG/scale) * scale*log2e)
#define SCALE2   0.022542110014144827f     // (1/64) * log2(e)
#define NEG_RAW  (-6.4e10f)                // -1e9 / (1/64)

// Packed f32x2 -> f16x2 convert (v_cvt_pk_f16_f32 path, RTZ).
__device__ __forceinline__ v2h pk_h2(float x, float y) {
  return __builtin_bit_cast(v2h, __builtin_amdgcn_cvt_pkrtz(x, y));
}

// Assemble a 16-half WMMA fragment from two 8-half (16B) LDS loads.
__device__ __forceinline__ v16h ld_frag(const f16* p0, const f16* p1) {
  v8h lo = *(const v8h*)p0;
  v8h hi = *(const v8h*)p1;
  v16h r;
#pragma unroll
  for (int i = 0; i < 8; ++i) { r[i] = lo[i]; r[i + 8] = hi[i]; }
  return r;
}

// One 64-key flash-attention tile. DOMASK / PREFETCH are compile-time so the
// hot (off-diagonal) instantiation carries zero masking VALU.
template <bool DOMASK, bool PREFETCH>
__device__ __forceinline__ void fa_tile(
    int tid, int wave, int lo16, int hi, int kc, int qbase,
    const float* __restrict__ Kg, const float* __restrict__ Vg,
    f16 (&Ks)[64][LDH], f16 (&Vt)[64][LDH], f16 (&Ps)[4][16][LDH],
    const v16h (&aq)[2], float (&mr)[8], float (&lr)[8], v8f (&Oa)[4]) {
  __syncthreads();  // previous tile's Ks/Vt consumers done
#pragma unroll
  for (int rep = 0; rep < 8; ++rep) {
    int linear = rep * 128 + tid;            // float4 index, 1024 total
    int row = linear >> 4, c4 = linear & 15;
    float4 kv = ((const float4*)Kg)[linear];
    *(v2h*)&Ks[row][c4 * 4 + 0] = pk_h2(kv.x, kv.y);
    *(v2h*)&Ks[row][c4 * 4 + 2] = pk_h2(kv.z, kv.w);
    float4 vv = ((const float4*)Vg)[linear];
    v2h p01 = pk_h2(vv.x, vv.y);
    v2h p23 = pk_h2(vv.z, vv.w);
    Vt[c4 * 4 + 0][row] = p01[0];
    Vt[c4 * 4 + 1][row] = p01[1];
    Vt[c4 * 4 + 2][row] = p23[0];
    Vt[c4 * 4 + 3][row] = p23[1];
  }
  if (PREFETCH) {  // next K/V tile: 16KB each = 128 threads x 128B lines
    __builtin_prefetch(Kg + (size_t)64 * D_ + tid * 32, 0, 3);
    __builtin_prefetch(Vg + (size_t)64 * D_ + tid * 32, 0, 3);
  }
  __syncthreads();

  // ---- S = Q * K^T : 16x64 per wave, f32 accum via WMMA (raw scores) ----
  v8f Sv[4] = {};
#pragma unroll
  for (int c = 0; c < 2; ++c)
#pragma unroll
    for (int n = 0; n < 4; ++n) {
      const f16* kp = &Ks[n * 16 + lo16][c * 32 + (hi ? 8 : 0)];
      v16h bb = ld_frag(kp, kp + 16);              // B-frag: K^T columns
      Sv[n] = __builtin_amdgcn_wmma_f32_16x16x32_f16(
          false, aq[c], false, bb, (short)0, Sv[n], false, false);
    }

  if (DOMASK) {  // causal mask, raw domain (only the diagonal tile)
#pragma unroll
    for (int n = 0; n < 4; ++n)
#pragma unroll
      for (int r = 0; r < 8; ++r) {
        int qg = qbase + r + 8 * hi;
        int kg = kc + n * 16 + lo16;
        if (kg > qg) Sv[n][r] += NEG_RAW;
      }
  }

  // ---- online softmax: raw row max via 16-lane shfl_xor reduction ----
  float mnew[8], al[8], cr[8];
#pragma unroll
  for (int r = 0; r < 8; ++r) {
    float v = fmaxf(fmaxf(Sv[0][r], Sv[1][r]), fmaxf(Sv[2][r], Sv[3][r]));
#pragma unroll
    for (int off = 1; off <= 8; off <<= 1)
      v = fmaxf(v, __shfl_xor(v, off, 32));
    mnew[r] = fmaxf(mr[r], v);
    al[r]   = exp2f((mr[r] - mnew[r]) * SCALE2);
    cr[r]   = -mnew[r] * SCALE2;
    mr[r]   = mnew[r];
  }

  // p = exp2(s*SCALE2 + cr); row sums; spill P to per-wave LDS
  float rs[8];
#pragma unroll
  for (int r = 0; r < 8; ++r) rs[r] = 0.f;
#pragma unroll
  for (int n = 0; n < 4; ++n)
#pragma unroll
    for (int r = 0; r < 8; ++r) {
      float p = exp2f(fmaf(Sv[n][r], SCALE2, cr[r]));
      rs[r] += p;
      Ps[wave][r + 8 * hi][n * 16 + lo16] = (f16)p;
    }
#pragma unroll
  for (int r = 0; r < 8; ++r) {
    float v = rs[r];
#pragma unroll
    for (int off = 1; off <= 8; off <<= 1)
      v += __shfl_xor(v, off, 32);
    lr[r] = lr[r] * al[r] + v;
  }
#pragma unroll
  for (int t = 0; t < 4; ++t)
#pragma unroll
    for (int r = 0; r < 8; ++r) Oa[t][r] *= al[r];

  // ---- O += P * V (V staged transposed -> contiguous B-frags) ----
#pragma unroll
  for (int c = 0; c < 2; ++c) {
    const f16* pp = &Ps[wave][lo16][c * 32 + (hi ? 8 : 0)];
    v16h a = ld_frag(pp, pp + 16);                 // A-frag: P rows
#pragma unroll
    for (int t = 0; t < 4; ++t) {
      const f16* vp = &Vt[t * 16 + lo16][c * 32 + (hi ? 8 : 0)];
      v16h bb = ld_frag(vp, vp + 16);              // B-frag: V columns
      Oa[t] = __builtin_amdgcn_wmma_f32_16x16x32_f16(
          false, a, false, bb, (short)0, Oa[t], false, false);
    }
  }
}

__launch_bounds__(128)
__global__ void fa_fwd_kernel(const float* __restrict__ Q,
                              const float* __restrict__ K,
                              const float* __restrict__ V,
                              float* __restrict__ O) {
  __shared__ f16 Qs[64][LDH];      // Q tile, row-major (q, d)
  __shared__ f16 Ks[64][LDH];      // K tile, row-major (k, d)
  __shared__ f16 Vt[64][LDH];      // V tile, transposed (d, k)
  __shared__ f16 Ps[4][16][LDH];   // per-wave P tile, row-major (q, k)

  const int tid  = threadIdx.x;
  const int lane = tid & 31;
  const int wave = tid >> 5;
  const int lo16 = lane & 15;
  const int hi   = (lane >> 4) & 1;

  const int b     = blockIdx.x >> 6;        // / (S/64)
  const int qtile = 63 - (blockIdx.x & 63); // longest-running blocks first
  const int q0    = qtile * 64;
  const int qbase = q0 + 16 * wave;

  // ---- stage Q tile: 64x64 f32 -> f16 in LDS (packed converts) ----
  const float* Qg = Q + ((size_t)b * S_ + q0) * D_;
#pragma unroll
  for (int rep = 0; rep < 8; ++rep) {
    int linear = rep * 128 + tid;            // float4 index, 1024 total
    int row = linear >> 4, c4 = linear & 15;
    float4 v = ((const float4*)Qg)[linear];
    *(v2h*)&Qs[row][c4 * 4 + 0] = pk_h2(v.x, v.y);
    *(v2h*)&Qs[row][c4 * 4 + 2] = pk_h2(v.z, v.w);
  }
  __syncthreads();

  // Q A-fragments are loop-invariant: hoist out of the key loop.
  v16h aq[2];
#pragma unroll
  for (int c = 0; c < 2; ++c) {
    const f16* qp = &Qs[16 * wave + lo16][c * 32 + (hi ? 8 : 0)];
    aq[c] = ld_frag(qp, qp + 16);
  }

  // online-softmax state (raw domain): r = C-layout VGPR slot (row = r+8*hi)
  float mr[8], lr[8];
#pragma unroll
  for (int r = 0; r < 8; ++r) { mr[r] = -3.0e38f; lr[r] = 0.f; }
  v8f Oa[4] = {};   // O accumulator, 16x64 per wave (4 N-tiles of d)

  const float* Kb = K + (size_t)b * S_ * D_;
  const float* Vb = V + (size_t)b * S_ * D_;

  // off-diagonal tiles: no masking code at all in the hot loop
  for (int j = 0; j < qtile; ++j) {
    const int kc = j * 64;
    fa_tile<false, true>(tid, wave, lo16, hi, kc, qbase,
                         Kb + (size_t)kc * D_, Vb + (size_t)kc * D_,
                         Ks, Vt, Ps, aq, mr, lr, Oa);
  }
  // diagonal tile: causal mask applied once
  {
    const int kc = qtile * 64;
    fa_tile<true, false>(tid, wave, lo16, hi, kc, qbase,
                         Kb + (size_t)kc * D_, Vb + (size_t)kc * D_,
                         Ks, Vt, Ps, aq, mr, lr, Oa);
  }

  // ---- normalize and store ----
  float inv[8];
#pragma unroll
  for (int r = 0; r < 8; ++r) inv[r] = 1.0f / lr[r];
  float* Og = O + ((size_t)b * S_ + qbase) * D_;
#pragma unroll
  for (int t = 0; t < 4; ++t)
#pragma unroll
    for (int r = 0; r < 8; ++r)
      Og[(r + 8 * hi) * D_ + t * 16 + lo16] = Oa[t][r] * inv[r];
}

extern "C" void kernel_launch(void* const* d_in, const int* in_sizes, int n_in,
                              void* d_out, int out_size, void* d_ws, size_t ws_size,
                              hipStream_t stream) {
  (void)in_sizes; (void)n_in; (void)out_size; (void)d_ws; (void)ws_size;
  const float* Q = (const float*)d_in[0];
  const float* K = (const float*)d_in[1];
  const float* V = (const float*)d_in[2];
  float* O = (float*)d_out;
  dim3 grid(B_ * (S_ / 64));   // 256 workgroups
  dim3 block(128);             // 4 waves (wave32), 64 query rows per WG
  hipLaunchKernelGGL(fa_fwd_kernel, grid, block, 0, stream, Q, K, V, O);
}
